// MutualCrossAttention_10969346474178
// MI455X (gfx1250) — matrix operand order repaired and checked by reading, not compiled, not run on hardware
//
#include <hip/hip_runtime.h>

constexpr int NB = 8;     // batch
constexpr int NS = 2048;  // seq len
constexpr int ND = 256;   // feature dim

typedef __attribute__((ext_vector_type(16))) _Float16 v16h;
typedef __attribute__((ext_vector_type(8)))  _Float16 v8h;
typedef __attribute__((ext_vector_type(8)))  float    v8f;
typedef __attribute__((ext_vector_type(4)))  unsigned int u32x4;
typedef __attribute__((ext_vector_type(8)))  int i32x8;
typedef __attribute__((ext_vector_type(4)))  int i32x4;

// ---- TDM availability / arity detection (device pass only) ----------------
#if defined(__AMDGCN__) && __has_builtin(__builtin_amdgcn_tensor_load_to_lds)
#define USE_TDM 1
#if __has_include(<hip/amd_detail/amd_gfx1250_TDM.h>)
#define TDM_6ARG 1  // therock/clang-23 flavor
#else
#define TDM_6ARG 0  // ROCm 7.2 / clang-22 flavor
#endif
#else
#define USE_TDM 0
#endif

static __device__ __forceinline__ v8f wmma16(v16h a, v16h b, v8f c) {
  // D = A(16x32 f16) * B(32x16 f16) + C(16x16 f32)
  return __builtin_amdgcn_wmma_f32_16x16x32_f16(false, a, false, b, (short)0, c,
                                                false, false);
}

// A-fragment (16x32 f16), row-major src, leading dim ld (halfs).
// lane r=lane&15 -> row r; hw=lane>>4 -> K {hw*8..+7} U {16+hw*8..+7}.
static __device__ __forceinline__ v16h load_a_frag(const _Float16* tile, int ld,
                                                   int hw, int ln) {
  const _Float16* p = tile + (size_t)ln * ld + hw * 8;
  v8h lo = *(const v8h*)(p);
  v8h hi = *(const v8h*)(p + 16);
  v16h r;
#pragma unroll
  for (int i = 0; i < 8; ++i) { r[i] = lo[i]; r[i + 8] = hi[i]; }
  return r;
}

// B-fragment (32x16 f16) from N-major storage: lane n reads
// K {hw*8..+7} U {16+hw*8..+7}. Two 16B loads (pitch may only be 16B aligned).
static __device__ __forceinline__ v16h load_b_frag(const _Float16* tile, int ld,
                                                   int hw, int ln) {
  const _Float16* p = tile + (size_t)ln * ld + hw * 8;
  v8h lo = *(const v8h*)(p);
  v8h hi = *(const v8h*)(p + 16);
  v16h r;
#pragma unroll
  for (int i = 0; i < 8; ++i) { r[i] = lo[i]; r[i + 8] = hi[i]; }
  return r;
}

#if USE_TDM
// 2D TDM tile load, D# packed per ISA 8.3/8.4. data_size = 2 bytes (f16).
static __device__ __forceinline__ void tdm_load_2d(
    unsigned long long gaddr, unsigned int lds_off, unsigned int dim0,
    unsigned int dim1, unsigned long long stride0, unsigned int pad_interval,
    unsigned int pad_amount) {
  u32x4 g0;
  g0[0] = 1u;                   // count=1, user mode, no gather
  g0[1] = lds_off;              // LDS byte address
  g0[2] = (unsigned int)gaddr;  // global_addr[31:0]
  g0[3] = ((unsigned int)(gaddr >> 32) & 0x01FFFFFFu) | 0x80000000u;  // type=2
  i32x8 g1;
  g1[0] = (int)((1u << 16) |            // data_size = 2 bytes
                (1u << 20) |            // pad_enable
                (pad_interval << 22) | (pad_amount << 25));
  g1[1] = (int)((dim0 & 0xFFFFu) << 16);                  // tensor_dim0 lo
  g1[2] = (int)((dim0 >> 16) | ((dim1 & 0xFFFFu) << 16)); // dim0 hi, dim1 lo
  g1[3] = (int)((dim1 >> 16) | (dim0 << 16));             // dim1 hi, tile_dim0
  g1[4] = (int)(dim1 & 0xFFFFu);                          // tile_dim1 (dim2=0)
  g1[5] = (int)(unsigned int)(stride0 & 0xFFFFFFFFu);     // dim0_stride lo
  g1[6] = (int)((stride0 >> 32) & 0xFFFFu);               // dim0_stride hi
  g1[7] = 0;
  i32x4 z4 = {0, 0, 0, 0};
#if TDM_6ARG
  i32x8 z8 = {0, 0, 0, 0, 0, 0, 0, 0};
  __builtin_amdgcn_tensor_load_to_lds(g0, g1, z4, z4, z8, 0);
#else
  __builtin_amdgcn_tensor_load_to_lds(g0, g1, z4, z4, 0);
#endif
}
#endif

// ---------------------------------------------------------------------------
// Kernel 1: fp32 -> f16 convert + 32x32 LDS tile transpose of x1/x2.
// ---------------------------------------------------------------------------
__global__ __launch_bounds__(256) void cvt_kernel(
    const float* __restrict__ x1, const float* __restrict__ x2,
    _Float16* __restrict__ x1h, _Float16* __restrict__ x2h,
    _Float16* __restrict__ x1T, _Float16* __restrict__ x2T) {
  __shared__ _Float16 tile[32][33];
  const int bz = blockIdx.z;
  const int which = bz & 1;
  const int bb = bz >> 1;
  const float* X = (which ? x2 : x1) + (size_t)bb * NS * ND;
  _Float16* H = (which ? x2h : x1h) + (size_t)bb * NS * ND;
  _Float16* T = (which ? x2T : x1T) + (size_t)bb * ND * NS;
  const int d0 = blockIdx.x * 32, s0 = blockIdx.y * 32;
  const int tx = threadIdx.x, ty = threadIdx.y;
#pragma unroll
  for (int k = 0; k < 4; ++k) {
    int s = s0 + ty + k * 8, d = d0 + tx;
    _Float16 h = (_Float16)X[(size_t)s * ND + d];
    H[(size_t)s * ND + d] = h;
    tile[ty + k * 8][tx] = h;
  }
  __syncthreads();
#pragma unroll
  for (int k = 0; k < 4; ++k) {
    int d = d0 + ty + k * 8, s = s0 + tx;
    T[(size_t)d * NS + s] = tile[tx][ty + k * 8];
  }
}

// ---------------------------------------------------------------------------
// Kernel 2: W fp32 [k][n] -> Wt f16 [n][k] (N-major for B-fragments).
// ---------------------------------------------------------------------------
__global__ __launch_bounds__(256) void wt_kernel(const float* __restrict__ W,
                                                 _Float16* __restrict__ Wt) {
  __shared__ _Float16 tile[32][33];
  const int n0 = blockIdx.x * 32, k0 = blockIdx.y * 32;
  const int tx = threadIdx.x, ty = threadIdx.y;
#pragma unroll
  for (int k = 0; k < 4; ++k) {
    int kk = k0 + ty + k * 8, n = n0 + tx;
    tile[ty + k * 8][tx] = (_Float16)W[(size_t)kk * ND + n];
  }
  __syncthreads();
#pragma unroll
  for (int k = 0; k < 4; ++k) {
    int n = n0 + ty + k * 8, kk = k0 + tx;
    Wt[(size_t)n * ND + kk] = tile[tx][ty + k * 8];
  }
}

// ---------------------------------------------------------------------------
// Kernel 3: q = (x1@W+b)*0.25, k = (x2@W+b)*0.25 via WMMA (scale folded so
// q.k score products come out pre-scaled by 1/sqrt(D)).
// ---------------------------------------------------------------------------
__global__ __launch_bounds__(256) void proj_kernel(
    const _Float16* __restrict__ x1h, const _Float16* __restrict__ x2h,
    const _Float16* __restrict__ Wt, const float* __restrict__ bias,
    _Float16* __restrict__ qh, _Float16* __restrict__ kh) {
  const int tid = threadIdx.x;
  const int wave = tid >> 5, lane = tid & 31;
  const int hw = lane >> 4, ln = lane & 15;
  int tileIdx = blockIdx.x * 8 + wave;
  const int which = (tileIdx >= 1024) ? 1 : 0;
  tileIdx &= 1023;
  const _Float16* X = which ? x2h : x1h;
  _Float16* Y = which ? kh : qh;
  const size_t row0 = (size_t)tileIdx * 16;

  v8f acc[16];
#pragma unroll
  for (int n = 0; n < 16; ++n)
#pragma unroll
    for (int r = 0; r < 8; ++r) acc[n][r] = 0.f;

#pragma unroll
  for (int kc = 0; kc < ND / 32; ++kc) {
    v16h a = load_a_frag(X + row0 * ND + kc * 32, ND, hw, ln);
#pragma unroll
    for (int n = 0; n < 16; ++n) {
      v16h b = load_b_frag(Wt + (size_t)(n * 16) * ND + kc * 32, ND, hw, ln);
      acc[n] = wmma16(a, b, acc[n]);
    }
  }
#pragma unroll
  for (int n = 0; n < 16; ++n) {
    float bv = bias[n * 16 + ln];
#pragma unroll
    for (int r = 0; r < 8; ++r) {
      float v = (acc[n][r] + bv) * 0.25f;
      Y[(row0 + r + hw * 8) * ND + n * 16 + ln] = (_Float16)v;
    }
  }
}

// ---------------------------------------------------------------------------
// Kernel 4: flash attention, both directions. TDM double-buffers the shared
// K tile (32x256) and V^T tile (256x32) into LDS; 8 waves each own a
// 16-query x 256-d output and consume the tiles via WMMA fragments.
// LDS pitches from TDM padding: K rows 272 halfs, V rows 40 halfs.
// ---------------------------------------------------------------------------
#define KBYTES 17408   // 32 rows * 544B
#define VBYTES 20480   // 256 rows * 80B
#define KOFF0 0
#define KOFF1 (KBYTES)
#define VOFF0 (2 * KBYTES)
#define VOFF1 (2 * KBYTES + VBYTES)
#define POFF  (2 * KBYTES + 2 * VBYTES)
#define SMEM_BYTES (POFF + 8 * 16 * 40 * 2)

__global__ __launch_bounds__(256) void fa_kernel(
    const _Float16* __restrict__ qh, const _Float16* __restrict__ kh,
    const _Float16* __restrict__ x1T, const _Float16* __restrict__ x2T,
    float* __restrict__ out) {
  __shared__ __align__(64) unsigned char smem[SMEM_BYTES];

  const int tid = threadIdx.x;
  const int wave = tid >> 5, lane = tid & 31;
  const int hw = lane >> 4, ln = lane & 15;
  const int bb = blockIdx.y;
  const int dir = blockIdx.z;
  const int row0 = (blockIdx.x * 8 + wave) * 16;

  const _Float16* Q = (dir ? kh : qh) + (size_t)bb * NS * ND;
  const _Float16* K = (dir ? qh : kh) + (size_t)bb * NS * ND;
  const _Float16* VT = (dir ? x1T : x2T) + (size_t)bb * ND * NS;
  float* O = out + ((size_t)bb * 2 * NS + (size_t)dir * NS + row0) * ND;
  _Float16* ps = (_Float16*)(smem + POFF) + wave * 16 * 40;

  float m[8], l[8];
  v8f acc[16];
#pragma unroll
  for (int r = 0; r < 8; ++r) { m[r] = -3.0e38f; l[r] = 0.f; }
#pragma unroll
  for (int n = 0; n < 16; ++n)
#pragma unroll
    for (int r = 0; r < 8; ++r) acc[n][r] = 0.f;

#if USE_TDM
  if (wave == 0) {  // prologue: DMA block 0 into buffer 0
    tdm_load_2d((unsigned long long)(const void*)K, KOFF0, ND, 32, ND, 6, 7);
    tdm_load_2d((unsigned long long)(const void*)VT, VOFF0, 32, ND, NS, 3, 2);
  }
#endif

  for (int i = 0; i < NS / 32; ++i) {
    const int t0 = i * 32;
    const int cur = i & 1;
#if USE_TDM
    if (wave == 0) {
      if (i + 1 < NS / 32) {  // DMA next block into the other buffer
        const int t1 = t0 + 32;
        tdm_load_2d((unsigned long long)(const void*)(K + (size_t)t1 * ND),
                    cur ? KOFF0 : KOFF1, ND, 32, ND, 6, 7);
        tdm_load_2d((unsigned long long)(const void*)(VT + t1),
                    cur ? VOFF0 : VOFF1, 32, ND, NS, 3, 2);
        __builtin_amdgcn_s_wait_tensorcnt(2);  // retire block i (in-order)
      } else {
        __builtin_amdgcn_s_wait_tensorcnt(0);
      }
    }
    __syncthreads();  // publish buffer `cur` to all waves
#else
    {  // fallback: cooperative synchronous staging
      _Float16* kb = (_Float16*)(smem + (cur ? KOFF1 : KOFF0));
      _Float16* vb = (_Float16*)(smem + (cur ? VOFF1 : VOFF0));
      for (int c = tid; c < 32 * 32; c += 256) {
        int r = c >> 5, col = (c & 31) * 8;
        *(v8h*)(kb + r * 272 + col) =
            *(const v8h*)(K + (size_t)(t0 + r) * ND + col);
      }
      for (int c = tid; c < 256 * 4; c += 256) {
        int r = c >> 2, col = (c & 3) * 8;
        *(v8h*)(vb + r * 40 + col) =
            *(const v8h*)(VT + (size_t)r * NS + t0 + col);
      }
      __syncthreads();
    }
#endif
    const _Float16* Kb = (const _Float16*)(smem + (cur ? KOFF1 : KOFF0));
    const _Float16* Vb = (const _Float16*)(smem + (cur ? VOFF1 : VOFF0));

    // ---- scores for 32 keys (pre-scaled; K tile from LDS) ----
    v8f s0 = {0.f, 0.f, 0.f, 0.f, 0.f, 0.f, 0.f, 0.f};
    v8f s1 = {0.f, 0.f, 0.f, 0.f, 0.f, 0.f, 0.f, 0.f};
#pragma unroll
    for (int kc = 0; kc < ND / 32; ++kc) {
      v16h a = load_a_frag(Q + (size_t)row0 * ND + kc * 32, ND, hw, ln);
      v16h b0 = load_b_frag(Kb + kc * 32, 272, hw, ln);
      v16h b1 = load_b_frag(Kb + 16 * 272 + kc * 32, 272, hw, ln);
      s0 = wmma16(a, b0, s0);
      s1 = wmma16(a, b1, s1);
    }
    // ---- online softmax; VGPR r holds row r+8*hw across 16 lanes ----
    float corr[8];
#pragma unroll
    for (int r = 0; r < 8; ++r) {
      float v = fmaxf(s0[r], s1[r]);
      v = fmaxf(v, __shfl_xor(v, 1));
      v = fmaxf(v, __shfl_xor(v, 2));
      v = fmaxf(v, __shfl_xor(v, 4));
      v = fmaxf(v, __shfl_xor(v, 8));
      float mn = fmaxf(m[r], v);
      corr[r] = __expf(m[r] - mn);
      m[r] = mn;
      float p0 = __expf(s0[r] - mn);
      float p1 = __expf(s1[r] - mn);
      float rs = p0 + p1;
      rs += __shfl_xor(rs, 1);
      rs += __shfl_xor(rs, 2);
      rs += __shfl_xor(rs, 4);
      rs += __shfl_xor(rs, 8);
      l[r] = l[r] * corr[r] + rs;
      int row = r + hw * 8;
      ps[row * 40 + ln] = (_Float16)p0;
      ps[row * 40 + 16 + ln] = (_Float16)p1;
    }
#pragma unroll
    for (int n = 0; n < 16; ++n)
#pragma unroll
      for (int r = 0; r < 8; ++r) acc[n][r] *= corr[r];
    // ---- P(16x32) * V(32x256) from LDS V^T tile ----
    v16h pa = load_a_frag(ps, 40, hw, ln);
#pragma unroll
    for (int n = 0; n < 16; ++n) {
      v16h vb = load_b_frag(Vb + (size_t)(n * 16) * 40, 40, hw, ln);
      acc[n] = wmma16(pa, vb, acc[n]);
    }
    __syncthreads();  // protect buffer `cur` before block i+2 DMA overwrites
  }
  // ---- epilogue ----
#pragma unroll
  for (int r = 0; r < 8; ++r) {
    float inv = 1.0f / l[r];
    int row = r + hw * 8;
#pragma unroll
    for (int n = 0; n < 16; ++n)
      O[(size_t)row * ND + n * 16 + ln] = acc[n][r] * inv;
  }
}

// ---------------------------------------------------------------------------
extern "C" void kernel_launch(void* const* d_in, const int* in_sizes, int n_in,
                              void* d_out, int out_size, void* d_ws,
                              size_t ws_size, hipStream_t stream) {
  (void)in_sizes; (void)n_in; (void)out_size; (void)ws_size;
  const float* x1 = (const float*)d_in[0];
  const float* x2 = (const float*)d_in[1];
  const float* W = (const float*)d_in[2];
  const float* bias = (const float*)d_in[3];
  float* out = (float*)d_out;

  const size_t NE = (size_t)NB * NS * ND;
  char* ws = (char*)d_ws;
  _Float16* x1h = (_Float16*)(ws + 0 * NE * 2);
  _Float16* x2h = (_Float16*)(ws + 1 * NE * 2);
  _Float16* x1T = (_Float16*)(ws + 2 * NE * 2);
  _Float16* x2T = (_Float16*)(ws + 3 * NE * 2);
  _Float16* qh  = (_Float16*)(ws + 4 * NE * 2);
  _Float16* kh  = (_Float16*)(ws + 5 * NE * 2);
  _Float16* Wt  = (_Float16*)(ws + 6 * NE * 2);

  cvt_kernel<<<dim3(ND / 32, NS / 32, NB * 2), dim3(32, 8), 0, stream>>>(
      x1, x2, x1h, x2h, x1T, x2T);
  wt_kernel<<<dim3(ND / 32, ND / 32, 1), dim3(32, 8), 0, stream>>>(W, Wt);
  proj_kernel<<<dim3(256, 1, 1), dim3(256, 1, 1), 0, stream>>>(x1h, x2h, Wt,
                                                               bias, qh, kh);
  fa_kernel<<<dim3(NS / 128, NB, 2), dim3(256, 1, 1), 0, stream>>>(qh, kh, x1T,
                                                                   x2T, out);
}
